// Dygre_10943576670534
// MI455X (gfx1250) — compile-verified, matrix operand order checked by reading, not run on hardware
//
#include <hip/hip_runtime.h>

typedef __attribute__((ext_vector_type(16))) _Float16 v16h;
typedef __attribute__((ext_vector_type(8)))  _Float16 v8h;
typedef __attribute__((ext_vector_type(8)))  float    v8f;
typedef __attribute__((ext_vector_type(4)))  float    fx4;   // native vec4 (AS-friendly)

// Global (address_space(1)) pointer views — forces global_load/global_store/
// global_atomic instead of flat_* (keeps DScnt out of the dependency chain).
typedef const float    __attribute__((address_space(1)))* gcf;
typedef float          __attribute__((address_space(1)))* gf;
typedef const _Float16 __attribute__((address_space(1)))* gch;
typedef _Float16       __attribute__((address_space(1)))* ghm;
typedef const fx4      __attribute__((address_space(1)))* gcf4;
typedef fx4            __attribute__((address_space(1)))* gf4;
typedef const v8h      __attribute__((address_space(1)))* gcv8h;
typedef const int      __attribute__((address_space(1)))* gci;

#define AS_GCF(p)  ((gcf)(uintptr_t)(p))
#define AS_GF(p)   ((gf)(uintptr_t)(p))
#define AS_GCH(p)  ((gch)(uintptr_t)(p))
#define AS_GHM(p)  ((ghm)(uintptr_t)(p))
#define AS_GCI(p)  ((gci)(uintptr_t)(p))

#define KDIM 128

__device__ __forceinline__ float sigm(float x) { return 1.0f / (1.0f + __expf(-x)); }

// ---------------------------------------------------------------------------
// Weight conversion kernels (f32 -> f16), run once per launch (tiny cost).
// ---------------------------------------------------------------------------
__global__ void k_cvt(const float* s_, _Float16* d_, int n) {
    int i = blockIdx.x * blockDim.x + threadIdx.x;
    gcf s = AS_GCF(s_);
    ghm d = AS_GHM(d_);
    if (i < n) d[i] = (_Float16)s[i];
}

// src is [128,128] (k-major rows, m = h @ W); dst = W^T as [n,k] row-major f16
__global__ void k_cvt_tr128(const float* s_, _Float16* d_) {
    int i = blockIdx.x * blockDim.x + threadIdx.x;
    gcf s = AS_GCF(s_);
    ghm d = AS_GHM(d_);
    if (i < 128 * 128) {
        int r = i >> 7, c = i & 127;
        d[i] = (_Float16)s[c * 128 + r];
    }
}

__global__ void k_zero4(float* p_, int n4) {
    int i = blockIdx.x * blockDim.x + threadIdx.x;
    gf4 p = (gf4)(uintptr_t)p_;
    if (i < n4) p[i] = (fx4){0.f, 0.f, 0.f, 0.f};
}

// ---------------------------------------------------------------------------
// WMMA GEMM:  C[M,OUT] = A1[M,128] * B1^T  (+ A2[M,128] * B2^T) (+ bias)
// B given as BT[OUT,128] row-major f16. One wave per 16-row x 64-col strip.
// A fragment (f32->f16 inline) reused across 4 column tiles (4 accumulators).
// ---------------------------------------------------------------------------
__global__ __launch_bounds__(256)
void k_gemm(const float* A1_, const _Float16* B1_,
            const float* A2_, const _Float16* B2_,
            float* C_, const float* bias_,
            int M, int OUT)
{
    int wave = (blockIdx.x * blockDim.x + threadIdx.x) >> 5;
    int lane = threadIdx.x & 31;
    int rowTiles  = M >> 4;
    int colGroups = OUT >> 6;
    if (wave >= rowTiles * colGroups) return;
    int rt = wave / colGroups;
    int cg = wave % colGroups;

    int lm = lane & 15;           // row (A) / col (B) within tile
    int lh = (lane >> 4) * 8;     // K half-offset per ISA 16-bit fragment layout
    int mrow = rt * 16 + lm;

    v8f acc[4] = {v8f{}, v8f{}, v8f{}, v8f{}};

    gcf Alist[2] = {AS_GCF(A1_), AS_GCF(A2_)};
    gch Blist[2] = {AS_GCH(B1_), AS_GCH(B2_)};
    int npass = (A2_ != nullptr) ? 2 : 1;

    for (int p = 0; p < npass; ++p) {
        gcf A  = Alist[p];
        gch BT = Blist[p];
        for (int kb = 0; kb < KDIM; kb += 32) {
            // A fragment: lane holds row mrow, K = {kb+lh..+7, kb+16+lh..+7}
            gcf ap = A + (size_t)mrow * KDIM + kb + lh;
            fx4 f0 = *(gcf4)(ap);
            fx4 f1 = *(gcf4)(ap + 4);
            fx4 f2 = *(gcf4)(ap + 16);
            fx4 f3 = *(gcf4)(ap + 20);
            v16h a;
            a[0]=(_Float16)f0.x;  a[1]=(_Float16)f0.y;  a[2]=(_Float16)f0.z;  a[3]=(_Float16)f0.w;
            a[4]=(_Float16)f1.x;  a[5]=(_Float16)f1.y;  a[6]=(_Float16)f1.z;  a[7]=(_Float16)f1.w;
            a[8]=(_Float16)f2.x;  a[9]=(_Float16)f2.y;  a[10]=(_Float16)f2.z; a[11]=(_Float16)f2.w;
            a[12]=(_Float16)f3.x; a[13]=(_Float16)f3.y; a[14]=(_Float16)f3.z; a[15]=(_Float16)f3.w;

            #pragma unroll
            for (int ct = 0; ct < 4; ++ct) {
                int ncol = cg * 64 + ct * 16 + lm;
                gch bp = BT + (size_t)ncol * KDIM + kb + lh;
                v8h b0 = *(gcv8h)(bp);
                v8h b1 = *(gcv8h)(bp + 16);
                v16h b = __builtin_shufflevector(b0, b1,
                          0,1,2,3,4,5,6,7,8,9,10,11,12,13,14,15);
                acc[ct] = __builtin_amdgcn_wmma_f32_16x16x32_f16(
                              false, a, false, b, (short)0, acc[ct], false, false);
            }
        }
    }

    // Store: C/D layout -> VGPR r, lanes 0-15: M=r, lanes 16-31: M=r+8; N=lane&15
    gf C = AS_GF(C_);
    gcf bias = AS_GCF(bias_);
    int rbase = rt * 16 + ((lane >> 4) * 8);
    #pragma unroll
    for (int ct = 0; ct < 4; ++ct) {
        int col = cg * 64 + ct * 16 + lm;
        float bv = bias_ ? bias[col] : 0.0f;
        #pragma unroll
        for (int r = 0; r < 8; ++r) {
            C[(size_t)(rbase + r) * OUT + col] = acc[ct][r] + bv;
        }
    }
}

// ---------------------------------------------------------------------------
// Edge scatter:  agg[dst] += m[src] * w   (1 lane = 1 edge x 4 channels)
// ---------------------------------------------------------------------------
__global__ __launch_bounds__(256)
void k_scatter(const float* m_, const int* src_, const int* dst_,
               const float* ew_, float* agg_, int E)
{
    long long gid = (long long)blockIdx.x * blockDim.x + threadIdx.x;
    int e = (int)(gid >> 5);
    if (e >= E) return;
    gcf m = AS_GCF(m_);
    gci src = AS_GCI(src_), dst = AS_GCI(dst_);
    gcf ew = AS_GCF(ew_);
    gf  agg = AS_GF(agg_);

    int c4 = ((int)gid & 31) * 4;
    int s = src[e], d = dst[e];
    float w = ew[e];
    fx4 v = *(gcf4)(m + (size_t)s * 128 + c4);
    gf base = agg + (size_t)d * 128 + c4;
    __hip_atomic_fetch_add(base + 0, v.x * w, __ATOMIC_RELAXED, __HIP_MEMORY_SCOPE_AGENT);
    __hip_atomic_fetch_add(base + 1, v.y * w, __ATOMIC_RELAXED, __HIP_MEMORY_SCOPE_AGENT);
    __hip_atomic_fetch_add(base + 2, v.z * w, __ATOMIC_RELAXED, __HIP_MEMORY_SCOPE_AGENT);
    __hip_atomic_fetch_add(base + 3, v.w * w, __ATOMIC_RELAXED, __HIP_MEMORY_SCOPE_AGENT);
}

// ---------------------------------------------------------------------------
// GRU cell gating: h_out = (1-z)*tanh(i_n + r*h_n) + z*h_in
// ---------------------------------------------------------------------------
__global__ __launch_bounds__(256)
void k_gru(const float* gi_, const float* gh_, const float* hin_, float* hout_,
           const float* bih_, const float* bhh_, int Nn)
{
    int idx = blockIdx.x * blockDim.x + threadIdx.x;
    if (idx >= Nn * 128) return;
    gcf gi = AS_GCF(gi_), gh = AS_GCF(gh_), hin = AS_GCF(hin_);
    gcf bih = AS_GCF(bih_), bhh = AS_GCF(bhh_);
    gf  hout = AS_GF(hout_);

    int n = idx >> 7, c = idx & 127;
    size_t g = (size_t)n * 384 + c;
    float ir = gi[g]       + bih[c],       hr = gh[g]       + bhh[c];
    float iz = gi[g + 128] + bih[c + 128], hz = gh[g + 128] + bhh[c + 128];
    float in_= gi[g + 256] + bih[c + 256], hn = gh[g + 256] + bhh[c + 256];
    float r = sigm(ir + hr);
    float z = sigm(iz + hz);
    float nc = tanhf(in_ + r * hn);
    hout[idx] = (1.0f - z) * nc + z * hin[idx];
}

// ---------------------------------------------------------------------------
// LSTM gating (gates already = h@Wih^T + hx@Whh^T, bias added here):
// writes h_new, c_new to d_out regions and relu(h_new) for the head GEMM.
// ---------------------------------------------------------------------------
__global__ __launch_bounds__(256)
void k_lstm(const float* gates_, const float* cx_,
            const float* bih_, const float* bhh_,
            float* hnew_, float* cnew_, float* relu_out_, int Nn)
{
    int idx = blockIdx.x * blockDim.x + threadIdx.x;
    if (idx >= Nn * 128) return;
    gcf gates = AS_GCF(gates_), cx = AS_GCF(cx_);
    gcf bih = AS_GCF(bih_), bhh = AS_GCF(bhh_);
    gf  hnew = AS_GF(hnew_), cnew = AS_GF(cnew_), relu_out = AS_GF(relu_out_);

    int n = idx >> 7, c = idx & 127;
    size_t g = (size_t)n * 512 + c;
    float ig = sigm(gates[g]       + bih[c]       + bhh[c]);
    float fg = sigm(gates[g + 128] + bih[c + 128] + bhh[c + 128]);
    float gg = tanhf(gates[g + 256] + bih[c + 256] + bhh[c + 256]);
    float og = sigm(gates[g + 384] + bih[c + 384] + bhh[c + 384]);
    float cn = fg * cx[idx] + ig * gg;
    float hn = og * tanhf(cn);
    hnew[idx] = hn;
    cnew[idx] = cn;
    relu_out[idx] = hn > 0.0f ? hn : 0.0f;
}

// ---------------------------------------------------------------------------
extern "C" void kernel_launch(void* const* d_in, const int* in_sizes, int n_in,
                              void* d_out, int out_size, void* d_ws, size_t ws_size,
                              hipStream_t stream)
{
    const float* x    = (const float*)d_in[0];
    const int*   ei   = (const int*)  d_in[1];
    const float* ew   = (const float*)d_in[2];
    const float* Hst  = (const float*)d_in[3];
    const float* Cst  = (const float*)d_in[4];
    const float* ggw  = (const float*)d_in[5];
    const float* gwih = (const float*)d_in[6];
    const float* gwhh = (const float*)d_in[7];
    const float* gbih = (const float*)d_in[8];
    const float* gbhh = (const float*)d_in[9];
    const float* lwih = (const float*)d_in[10];
    const float* lwhh = (const float*)d_in[11];
    const float* lbih = (const float*)d_in[12];
    const float* lbhh = (const float*)d_in[13];
    const float* linw = (const float*)d_in[14];
    const float* linb = (const float*)d_in[15];

    const int Nn = in_sizes[0] / 128;        // 50000 (multiple of 16)
    const int E  = in_sizes[2];              // 800000
    const int* srcIdx = ei;
    const int* dstIdx = ei + E;

    float* out      = (float*)d_out;                     // [N,64]
    float* hnew_out = out + (size_t)Nn * 64;             // [N,128]
    float* cnew_out = hnew_out + (size_t)Nn * 128;       // [N,128]

    float* ws  = (float*)d_ws;
    float* h   = ws;                                     // [N,128]
    float* m   = h   + (size_t)Nn * 128;                 // [N,128] (reused as relu buf)
    float* agg = m   + (size_t)Nn * 128;                 // [N,128]
    float* big = agg + (size_t)Nn * 128;                 // [N,768]
    float* gi  = big;                                    // [N,384]
    float* gh  = big + (size_t)Nn * 384;                 // [N,384]
    float* lstm_gates = big;                             // [N,512]

    _Float16* wp    = (_Float16*)(big + (size_t)Nn * 768);
    _Float16* wg0   = wp;                //  128*128
    _Float16* wg1   = wg0   + 16384;     //  128*128
    _Float16* cwih  = wg1   + 16384;     //  384*128
    _Float16* cwhh  = cwih  + 49152;     //  384*128
    _Float16* clwih = cwhh  + 49152;     //  512*128
    _Float16* clwhh = clwih + 65536;     //  512*128
    _Float16* clw   = clwhh + 65536;     //   64*128

    const int T = 256;
    auto blk = [](long long n, int t) { return (unsigned)((n + t - 1) / t); };

    // --- weight conversion -------------------------------------------------
    k_cvt_tr128<<<blk(16384, T), T, 0, stream>>>(ggw, wg0);
    k_cvt_tr128<<<blk(16384, T), T, 0, stream>>>(ggw + 16384, wg1);
    k_cvt<<<blk(49152, T), T, 0, stream>>>(gwih, cwih, 49152);
    k_cvt<<<blk(49152, T), T, 0, stream>>>(gwhh, cwhh, 49152);
    k_cvt<<<blk(65536, T), T, 0, stream>>>(lwih, clwih, 65536);
    k_cvt<<<blk(65536, T), T, 0, stream>>>(lwhh, clwhh, 65536);
    k_cvt<<<blk(8192,  T), T, 0, stream>>>(linw, clw, 8192);

    const int rowTiles = Nn / 16;
    auto gemm = [&](const float* A1, const _Float16* B1,
                    const float* A2, const _Float16* B2,
                    float* C, const float* bias, int OUT) {
        long long waves = (long long)rowTiles * (OUT / 64);
        k_gemm<<<blk(waves * 32, T), T, 0, stream>>>(A1, B1, A2, B2, C, bias, Nn, OUT);
    };

    const long long NC  = (long long)Nn * 128;
    const unsigned  ewB = blk(NC, T);

    // --- GatedGraphConv layers --------------------------------------------
    for (int layer = 0; layer < 2; ++layer) {
        const float* hcur = (layer == 0) ? x : h;
        const _Float16* wg = (layer == 0) ? wg0 : wg1;

        gemm(hcur, wg, nullptr, nullptr, m, nullptr, 128);        // m = h @ W
        k_zero4<<<blk(NC / 4, T), T, 0, stream>>>(agg, (int)(NC / 4));
        k_scatter<<<blk((long long)E * 32, T), T, 0, stream>>>(m, srcIdx, dstIdx, ew, agg, E);
        gemm(agg,  cwih, nullptr, nullptr, gi, nullptr, 384);     // gi = agg @ Wih^T
        gemm(hcur, cwhh, nullptr, nullptr, gh, nullptr, 384);     // gh = h   @ Whh^T
        k_gru<<<ewB, T, 0, stream>>>(gi, gh, hcur, h, gbih, gbhh, Nn);
    }

    // --- LSTM (fused dual GEMM) + head ------------------------------------
    gemm(h, clwih, Hst, clwhh, lstm_gates, nullptr, 512);
    k_lstm<<<ewB, T, 0, stream>>>(lstm_gates, Cst, lbih, lbhh,
                                  hnew_out, cnew_out, m, Nn);
    gemm(m, clw, nullptr, nullptr, out, linb, 64);                // out = relu(h)@lin^T + b
}
